// GraphUNet_LSTM_87797721464867
// MI455X (gfx1250) — matrix operand order repaired and checked by reading, not compiled, not run on hardware
//
#include <hip/hip_runtime.h>
#include <hip/hip_bf16.h>
#include <math.h>

// ---------------------------------------------------------------------------
// GraphUNet+LSTM for MI455X (gfx1250).  All dense matmuls routed through
// v_wmma_f32_16x16x32_bf16 (bf16 A/B, f32 accumulate).  Adjacency entries are
// small integer counts -> exact in bf16; activations get RNE bf16 rounding.
// LSTM recurrence keeps Whh resident in LDS (256KB of the 320KB WGP LDS).
// GEMM double-buffers LDS tiles: the A tile for step k+1 is fetched with
// gfx1250 async global->LDS b128 copies (ASYNCcnt) while step k's WMMAs run.
// ---------------------------------------------------------------------------

#define N0   4096
#define TT   12
#define CIN  32
#define HDIM 128
#define COUT 32
#define GDIM 512   // 4*HDIM

typedef __bf16 bf16_t;
typedef bf16_t v16bf __attribute__((ext_vector_type(16)));
typedef float  v8f   __attribute__((ext_vector_type(8)));

// ---- gfx1250 async global->LDS path:
//      void(v4i AS1* src, v4i AS3* dst, imm int offset, imm int cpol) ----
#if defined(__has_builtin)
#if __has_builtin(__builtin_amdgcn_global_load_async_to_lds_b128)
#define HAVE_ASYNC_LDS 1
#endif
#endif
#ifndef HAVE_ASYNC_LDS
#define HAVE_ASYNC_LDS 0
#endif

#if HAVE_ASYNC_LDS
typedef int v4i_t __attribute__((vector_size(16)));
typedef __attribute__((address_space(1))) v4i_t* gas1_t;
typedef __attribute__((address_space(3))) v4i_t* las3_t;
__device__ __forceinline__ gas1_t as_global(const void* p) {
  return (gas1_t)(unsigned long long)p;
}
__device__ __forceinline__ las3_t as_lds(const void* p) {
  // flat pointers into LDS carry the LDS byte offset in their low 32 bits
  return (las3_t)(unsigned)(unsigned long long)p;
}
__device__ __forceinline__ void wait_async0() {
#if __has_builtin(__builtin_amdgcn_s_wait_asynccnt)
  __builtin_amdgcn_s_wait_asynccnt(0);
#else
  asm volatile("s_wait_asynccnt 0" ::: "memory");
#endif
}
#endif

// ---------------- elementwise / utility kernels ----------------

__global__ void fill0_k(float* p, size_t n) {
  size_t i = blockIdx.x * (size_t)blockDim.x + threadIdx.x;
  size_t st = (size_t)gridDim.x * blockDim.x;
  for (; i < n; i += st) p[i] = 0.0f;
}

__global__ void cast_bf16_k(const float* s, bf16_t* d, size_t n) {
  size_t i = blockIdx.x * (size_t)blockDim.x + threadIdx.x;
  size_t st = (size_t)gridDim.x * blockDim.x;
  for (; i < n; i += st) d[i] = (bf16_t)s[i];
}

// src[rows][cols] -> dst[cols][rows]  (used for Wih -> Wih^T)
__global__ void transpose_cast_k(const float* s, bf16_t* d, int rows, int cols) {
  size_t n = (size_t)rows * cols;
  size_t i = blockIdx.x * (size_t)blockDim.x + threadIdx.x;
  size_t st = (size_t)gridDim.x * blockDim.x;
  for (; i < n; i += st) {
    int r = (int)(i / cols), c = (int)(i % cols);
    d[(size_t)c * rows + r] = (bf16_t)s[i];
  }
}

// x[N0][TT][CIN] -> bf16 x_last[N0][CIN]
__global__ void xlast_cast_k(const float* x, bf16_t* d) {
  size_t n = (size_t)N0 * CIN;
  size_t i = blockIdx.x * (size_t)blockDim.x + threadIdx.x;
  size_t st = (size_t)gridDim.x * blockDim.x;
  for (; i < n; i += st) {
    int r = (int)(i / CIN), c = (int)(i % CIN);
    d[i] = (bf16_t)x[((size_t)r * TT + (TT - 1)) * CIN + c];
  }
}

// edge_index[2][E]; M[dst][src] += 1 for non-self-loops
__global__ void build_adj_k(const int* ei, int E, float* M) {
  int e = blockIdx.x * blockDim.x + threadIdx.x;
  if (e >= E) return;
  int s = ei[e], d = ei[E + e];
  if (s != d) atomicAdd(&M[(size_t)d * N0 + s], 1.0f);
}

// dis[r] = (rowsum(M)+2)^-0.5
__global__ void degdis_k(const float* M, int n, float* dis) {
  int r = blockIdx.x * blockDim.x + threadIdx.x;
  if (r >= n) return;
  float s = 0.0f;
  const float4* row = (const float4*)(M + (size_t)r * n);
  int n4 = n >> 2;
  for (int c = 0; c < n4; ++c) {
    float4 v = row[c];
    s += v.x + v.y + v.z + v.w;
  }
  s += 2.0f;
  dis[r] = (s > 0.0f) ? rsqrtf(s) : 0.0f;
}

// Ms = M with diagonal forced to 1 (bf16 -> bf16)
__global__ void set_diag1_k(const bf16_t* s, bf16_t* d, int n) {
  size_t tot = (size_t)n * n;
  size_t i = blockIdx.x * (size_t)blockDim.x + threadIdx.x;
  size_t st = (size_t)gridDim.x * blockDim.x;
  for (; i < tot; i += st) {
    int r = (int)(i / n), c = (int)(i % n);
    d[i] = (r == c) ? (bf16_t)1.0f : s[i];
  }
}

__global__ void rowscale_k(const float* X, const float* dis, float* Z, int n, int C) {
  size_t tot = (size_t)n * C;
  size_t i = blockIdx.x * (size_t)blockDim.x + threadIdx.x;
  size_t st = (size_t)gridDim.x * blockDim.x;
  for (; i < tot; i += st) Z[i] = dis[i / C] * X[i];
}

// out = dis*(Y + 2Z) + b  (+ optional relu)
__global__ void gcn_epi_k(const float* Y, const float* Z, const float* dis,
                          const float* b, float* o, int n, int C, int dorelu) {
  size_t tot = (size_t)n * C;
  size_t i = blockIdx.x * (size_t)blockDim.x + threadIdx.x;
  size_t st = (size_t)gridDim.x * blockDim.x;
  for (; i < tot; i += st) {
    int r = (int)(i / C), c = (int)(i % C);
    float v = dis[r] * (Y[i] + 2.0f * Z[i]) + b[c];
    if (dorelu) v = fmaxf(v, 0.0f);
    o[i] = v;
  }
}

__global__ void wnorm_k(const float* w, float* invn) {
  __shared__ float red[HDIM];
  int t = threadIdx.x;
  float v = w[t];
  red[t] = v * v;
  __syncthreads();
  for (int s = HDIM / 2; s > 0; s >>= 1) {
    if (t < s) red[t] += red[t + s];
    __syncthreads();
  }
  if (t == 0) *invn = rsqrtf(red[0]);
}

__global__ void score_k(const float* h, const float* w, const float* invn,
                        float* sc, int n) {
  int r = blockIdx.x * blockDim.x + threadIdx.x;
  if (r >= n) return;
  float a = 0.0f;
  const float* row = h + (size_t)r * HDIM;
  for (int c = 0; c < HDIM; ++c) a += row[c] * w[c];
  sc[r] = tanhf(a * (*invn));
}

// full bitonic sort (desc by score, ties: lower index first); emit top-k
__global__ void topk_sort_k(const float* sc, int n, int k, int* perm, float* ss) {
  extern __shared__ float tsm[];
  float* keys = tsm;
  int*   vals = (int*)(tsm + n);
  int tid = threadIdx.x, bsz = blockDim.x;
  for (int i = tid; i < n; i += bsz) { keys[i] = sc[i]; vals[i] = i; }
  __syncthreads();
  for (int size = 2; size <= n; size <<= 1) {
    for (int stride = size >> 1; stride > 0; stride >>= 1) {
      for (int i = tid; i < n; i += bsz) {
        int j = i ^ stride;
        if (j > i) {
          bool dirFwd = ((i & size) == 0);
          float ki = keys[i], kj = keys[j];
          int vi = vals[i], vj = vals[j];
          bool iBefore = (ki > kj) || (ki == kj && vi < vj);
          if (iBefore != dirFwd) {
            keys[i] = kj; keys[j] = ki;
            vals[i] = vj; vals[j] = vi;
          }
        }
      }
      __syncthreads();
    }
  }
  for (int i = tid; i < k; i += bsz) { perm[i] = vals[i]; ss[i] = keys[i]; }
}

__global__ void gather_x_k(const float* h, const int* perm, const float* ss,
                           float* o, int k) {
  size_t tot = (size_t)k * HDIM;
  size_t i = blockIdx.x * (size_t)blockDim.x + threadIdx.x;
  size_t st = (size_t)gridDim.x * blockDim.x;
  for (; i < tot; i += st) {
    int r = (int)(i / HDIM), c = (int)(i % HDIM);
    o[i] = h[(size_t)perm[r] * HDIM + c] * ss[r];
  }
}

// Mp[r][c] = (r==c) ? 0 : M2[perm[r]][perm[c]]   (diag removal fused)
__global__ void gather_M_k(const float* M2, const int* perm, int nold, int k,
                           float* Mp) {
  size_t tot = (size_t)k * k;
  size_t i = blockIdx.x * (size_t)blockDim.x + threadIdx.x;
  size_t st = (size_t)gridDim.x * blockDim.x;
  for (; i < tot; i += st) {
    int r = (int)(i / k), c = (int)(i % k);
    Mp[i] = (r == c) ? 0.0f : M2[(size_t)perm[r] * nold + perm[c]];
  }
}

// dst[perm[r]][c] += h[r][c]  (perm rows distinct -> no atomics needed)
__global__ void scatter_add_k(float* dst, const float* h, const int* perm,
                              int k, int C) {
  size_t tot = (size_t)k * C;
  size_t i = blockIdx.x * (size_t)blockDim.x + threadIdx.x;
  size_t st = (size_t)gridDim.x * blockDim.x;
  for (; i < tot; i += st) {
    int r = (int)(i / C), c = (int)(i % C);
    dst[(size_t)perm[r] * C + c] += h[i];
  }
}

// ---------------- LSTM sequential scan (one workgroup, Whh in LDS) --------
// 512 threads: thread j owns gate row j.  Whh fp32 = 256KB in the 320KB LDS.
__global__ void lstm_scan_k(const float* xg, const float* Whh,
                            const float* bih, const float* bhh,
                            float* out, int n) {
  extern __shared__ float lsm[];
  float* Wh = lsm;                       // GDIM*HDIM
  float* hs = Wh + GDIM * HDIM;          // HDIM
  float* cs = hs + HDIM;                 // HDIM
  float* gs = cs + HDIM;                 // GDIM
  int tid = threadIdx.x;                 // 512
  for (int i = tid; i < GDIM * HDIM; i += GDIM) Wh[i] = Whh[i];
  if (tid < HDIM) { hs[tid] = 0.0f; cs[tid] = 0.0f; }
  __syncthreads();
  float bsum = bih[tid] + bhh[tid];
  const float4* w4 = (const float4*)&Wh[tid * HDIM];
  const float4* h4 = (const float4*)hs;
  for (int t = 0; t < n; ++t) {
    float acc = xg[(size_t)t * GDIM + tid] + bsum;
#pragma unroll 8
    for (int kk = 0; kk < HDIM / 4; ++kk) {
      float4 w = w4[kk], h = h4[kk];
      acc += w.x * h.x + w.y * h.y + w.z * h.z + w.w * h.w;
    }
    gs[tid] = acc;
    __syncthreads();
    if (tid < HDIM) {
      float i_ = 1.0f / (1.0f + expf(-gs[tid]));
      float f_ = 1.0f / (1.0f + expf(-gs[HDIM + tid]));
      float g_ = tanhf(gs[2 * HDIM + tid]);
      float o_ = 1.0f / (1.0f + expf(-gs[3 * HDIM + tid]));
      float c = f_ * cs[tid] + i_ * g_;
      float h = o_ * tanhf(c);
      cs[tid] = c; hs[tid] = h;
      out[(size_t)t * HDIM + tid] = h;
    }
    __syncthreads();
  }
}

// ---------------- bf16 WMMA GEMM:  C[M,N] = A[M,K] @ B[K,N]  ---------------
// Block: 256 thr = 8 waves; each wave owns a 32x32 C tile (4 accumulators,
// A/B fragments reused across two WMMAs each).  LDS tiles are DOUBLE-BUFFERED:
// step k+1's A tile is fetched with async global->LDS b128 copies (ASYNCcnt)
// while step k's WMMAs execute; B tile is staged transposed ([col][k]) so
// fragments are contiguous b128 LDS loads.
// Fragment layouts per CDNA5 ISA 7.12.2 (wave32):
//   A 16x32 bf16: lane<16 -> row=lane, K {0..7,16..23}; lane>=16 -> K {8..15,24..31}
//   B 32x16 bf16: lane<16 -> col=lane, K {0..7,16..23}; lane>=16 -> K {8..15,24..31}
//   C 16x16 f32 : vgpr r, lanes 0..15 -> m=r, lanes 16..31 -> m=8+r
#define BM 256
#define BN 32
#define BK 32
#define AQ_PER_BUF (BM * BK / 8)   // 1024 uint4 slots
#define BQ_PER_BUF (BN * BK / 8)   // 128 uint4 slots
__global__ __launch_bounds__(256) void gemm_bf16_k(const bf16_t* __restrict__ A,
                                                   const bf16_t* __restrict__ B,
                                                   float* __restrict__ C,
                                                   int M, int N, int K) {
  __shared__ __align__(16) bf16_t As[2 * BM * BK];   // 2 x 16KB, row-major
  __shared__ __align__(16) bf16_t BsT[2 * BN * BK];  // 2 x 2KB, [col][k]
  int tid = threadIdx.x;
  int wave = tid >> 5, lane = tid & 31;
  int hi = lane >> 4, l16 = lane & 15;
  int blockRow = blockIdx.y * BM;
  int blockCol = blockIdx.x * BN;
  uint4* AsQ = (uint4*)As;
  const uint4* BsQ = (const uint4*)BsT;

  auto stage = [&](int k0, int buf) {
    // A tile: 256x32 bf16 = 16KB -> 4 x b128 per thread (async to LDS)
#pragma unroll
    for (int s = 0; s < 4; ++s) {
      int q = tid + s * 256;
      const bf16_t* g = A + (size_t)(blockRow + (q >> 2)) * K + k0 + (q & 3) * 8;
#if HAVE_ASYNC_LDS
      __builtin_amdgcn_global_load_async_to_lds_b128(
          as_global(g), as_lds(&AsQ[buf * AQ_PER_BUF + q]), 0, 0);
#else
      AsQ[buf * AQ_PER_BUF + q] = *(const uint4*)g;
#endif
    }
    // B tile transposed (32x32 bf16): waves 0..3, b128 global loads
    if (tid < 128) {
      int r = tid >> 2, c0 = (tid & 3) * 8;
      union { uint4 q; bf16_t e[8]; } u;
      u.q = *(const uint4*)(B + (size_t)(k0 + r) * N + blockCol + c0);
      bf16_t* bs = &BsT[buf * BN * BK];
#pragma unroll
      for (int t = 0; t < 8; ++t) bs[(c0 + t) * BK + r] = u.e[t];
    }
  };

  v8f acc00 = {}, acc01 = {}, acc10 = {}, acc11 = {};
  stage(0, 0);
#if HAVE_ASYNC_LDS
  wait_async0();
#endif
  __syncthreads();

  int nb = K / BK;
  for (int ib = 0; ib < nb; ++ib) {
    int cur = ib & 1;
    if (ib + 1 < nb) {
      stage((ib + 1) * BK, cur ^ 1);
      __builtin_prefetch(A + (size_t)(blockRow + (tid >> 1)) * K + (ib + 2) * BK, 0, 1);
    }
    // ---- fragments (b128 LDS loads) + 4 WMMA on the current buffer ----
    const uint4* aq = &AsQ[cur * AQ_PER_BUF];
    const uint4* bq = &BsQ[cur * BQ_PER_BUF];
    int row0 = wave * 32 + l16, row1 = row0 + 16;
    union { v16bf v; uint4 q[2]; } af0, af1, bf0, bf1;
    af0.q[0] = aq[row0 * 4 + hi];      af0.q[1] = aq[row0 * 4 + 2 + hi];
    af1.q[0] = aq[row1 * 4 + hi];      af1.q[1] = aq[row1 * 4 + 2 + hi];
    bf0.q[0] = bq[l16 * 4 + hi];   bf0.q[1] = bq[l16 * 4 + 2 + hi];
    bf1.q[0] = bq[(16 + l16) * 4 + hi];
    bf1.q[1] = bq[(16 + l16) * 4 + 2 + hi];
    acc00 = __builtin_amdgcn_wmma_f32_16x16x32_bf16(false, af0.v, false, bf0.v,
                                                    (short)0, acc00, false, false);
    acc01 = __builtin_amdgcn_wmma_f32_16x16x32_bf16(false, af0.v, false, bf1.v,
                                                    (short)0, acc01, false, false);
    acc10 = __builtin_amdgcn_wmma_f32_16x16x32_bf16(false, af1.v, false, bf0.v,
                                                    (short)0, acc10, false, false);
    acc11 = __builtin_amdgcn_wmma_f32_16x16x32_bf16(false, af1.v, false, bf1.v,
                                                    (short)0, acc11, false, false);
    if (ib + 1 < nb) {
#if HAVE_ASYNC_LDS
      wait_async0();   // drain async copies for the next buffer
#endif
      __syncthreads(); // next buffer staged; current buffer free for reuse
    }
  }
  int rb0 = blockRow + wave * 32 + hi * 8;
  int rb1 = rb0 + 16;
  int c0 = blockCol + l16, c1 = blockCol + 16 + l16;
#pragma unroll
  for (int r = 0; r < 8; ++r) {
    C[(size_t)(rb0 + r) * N + c0] = acc00[r];
    C[(size_t)(rb0 + r) * N + c1] = acc01[r];
    C[(size_t)(rb1 + r) * N + c0] = acc10[r];
    C[(size_t)(rb1 + r) * N + c1] = acc11[r];
  }
}

// ---------------------------------------------------------------------------

static inline dim3 g1(size_t work, int b) {
  size_t g = (work + b - 1) / b;
  if (g > 65535u * 16u) g = 65535u * 16u;
  return dim3((unsigned)g);
}

extern "C" void kernel_launch(void* const* d_in, const int* in_sizes, int n_in,
                              void* d_out, int out_size, void* d_ws, size_t ws_size,
                              hipStream_t stream) {
  (void)n_in; (void)out_size; (void)ws_size;
  const float* x    = (const float*)d_in[0];
  const int*   edge = (const int*)d_in[1];
  const int    E    = in_sizes[1] / 2;
  const float* down_W[4]; const float* down_b[4];
  const float* Wih[4]; const float* Whh[4]; const float* bih[4]; const float* bhh[4];
  const float* pool_w[3]; const float* up_W[3]; const float* up_b[3];
  for (int l = 0; l < 4; ++l) {
    down_W[l] = (const float*)d_in[2 + l];
    down_b[l] = (const float*)d_in[6 + l];
    Wih[l]    = (const float*)d_in[10 + l];
    Whh[l]    = (const float*)d_in[14 + l];
    bih[l]    = (const float*)d_in[18 + l];
    bhh[l]    = (const float*)d_in[22 + l];
  }
  for (int l = 0; l < 3; ++l) {
    pool_w[l] = (const float*)d_in[26 + l];
    up_W[l]   = (const float*)d_in[29 + l];
    up_b[l]   = (const float*)d_in[32 + l];
  }

  const int nlev[4] = {4096, 2048, 1024, 512};

  // ---- workspace layout (bump allocator, 256B aligned) ----
  char* base = (char*)d_ws; size_t off = 0;
  auto alloc = [&](size_t b) -> void* {
    void* p = base + off; off = (off + b + 255) & ~(size_t)255; return p;
  };
  float*  Af   = (float*)alloc((size_t)N0 * N0 * 4);          // M0 f32 / M2 scratch
  float*  Bf   = (float*)alloc((size_t)2048 * 2048 * 4);      // pooled M f32 scratch
  bf16_t* Mb[4];
  for (int l = 0; l < 4; ++l) Mb[l] = (bf16_t*)alloc((size_t)nlev[l] * nlev[l] * 2);
  bf16_t* Msb  = (bf16_t*)alloc((size_t)N0 * N0 * 2);         // diag-1 adjacency
  float*  dis[4];
  for (int l = 0; l < 4; ++l) dis[l] = (float*)alloc((size_t)nlev[l] * 4);
  bf16_t* dwb[4];
  dwb[0] = (bf16_t*)alloc((size_t)CIN * HDIM * 2);
  for (int l = 1; l < 4; ++l) dwb[l] = (bf16_t*)alloc((size_t)HDIM * HDIM * 2);
  bf16_t* uwb[3];
  uwb[0] = (bf16_t*)alloc((size_t)HDIM * HDIM * 2);
  uwb[1] = (bf16_t*)alloc((size_t)HDIM * HDIM * 2);
  uwb[2] = (bf16_t*)alloc((size_t)HDIM * COUT * 2);
  bf16_t* wihT[4];
  for (int l = 0; l < 4; ++l) wihT[l] = (bf16_t*)alloc((size_t)HDIM * GDIM * 2);
  bf16_t* xlb  = (bf16_t*)alloc((size_t)N0 * CIN * 2);
  float*  XW   = (float*)alloc((size_t)N0 * HDIM * 4);
  float*  Zf   = (float*)alloc((size_t)N0 * HDIM * 4);
  bf16_t* Zb   = (bf16_t*)alloc((size_t)N0 * HDIM * 2);
  float*  Yf   = (float*)alloc((size_t)N0 * HDIM * 4);
  float*  hA   = (float*)alloc((size_t)N0 * HDIM * 4);
  float*  hB   = (float*)alloc((size_t)N0 * HDIM * 4);
  bf16_t* hb   = (bf16_t*)alloc((size_t)N0 * HDIM * 2);
  float*  XG   = (float*)alloc((size_t)N0 * GDIM * 4);
  float*  xs0  = (float*)alloc((size_t)4096 * HDIM * 4);
  float*  xs1  = (float*)alloc((size_t)2048 * HDIM * 4);
  float*  xs2  = (float*)alloc((size_t)1024 * HDIM * 4);
  int*    perm[3];
  perm[0] = (int*)alloc(2048 * 4);
  perm[1] = (int*)alloc(1024 * 4);
  perm[2] = (int*)alloc(512 * 4);
  float*  sscore = (float*)alloc(2048 * 4);
  float*  scoreb = (float*)alloc(4096 * 4);
  float*  invn   = (float*)alloc(256);
  float*  inbuf  = (float*)alloc((size_t)N0 * HDIM * 4);

  auto gemm = [&](const bf16_t* A, const bf16_t* B, float* C, int M, int N, int K) {
    dim3 grid(N / BN, M / BM);
    gemm_bf16_k<<<grid, 256, 0, stream>>>(A, B, C, M, N, K);
  };
  auto cast = [&](const float* s, bf16_t* d, size_t n) {
    cast_bf16_k<<<g1(n, 256), 256, 0, stream>>>(s, d, n);
  };
  // gcn: out = dis*(M@(dis*(in@W)) + 2*dis*(in@W)) + b   (hin already bf16)
  auto gcn = [&](const bf16_t* Mbm, const float* disv, int n, const bf16_t* hinb,
                 int Kin, const bf16_t* Wb, int Cout, const float* bias,
                 bool relu, float* outp) {
    gemm(hinb, Wb, XW, n, Cout, Kin);
    rowscale_k<<<g1((size_t)n * Cout, 256), 256, 0, stream>>>(XW, disv, Zf, n, Cout);
    cast(Zf, Zb, (size_t)n * Cout);
    gemm(Mbm, Zb, Yf, n, Cout, n);
    gcn_epi_k<<<g1((size_t)n * Cout, 256), 256, 0, stream>>>(Yf, Zf, disv, bias,
                                                             outp, n, Cout, relu ? 1 : 0);
  };
  const size_t LSTM_LDS = (size_t)(GDIM * HDIM + 2 * HDIM + GDIM) * sizeof(float);
  auto dolstm = [&](const float* hin, int n, int l, float* outp) {
    cast(hin, hb, (size_t)n * HDIM);
    gemm(hb, wihT[l], XG, n, GDIM, HDIM);
    lstm_scan_k<<<1, GDIM, LSTM_LDS, stream>>>(XG, Whh[l], bih[l], bhh[l], outp, n);
  };

  // ---- weight precasting (every call; deterministic) ----
  cast(down_W[0], dwb[0], (size_t)CIN * HDIM);
  for (int l = 1; l < 4; ++l) cast(down_W[l], dwb[l], (size_t)HDIM * HDIM);
  cast(up_W[0], uwb[0], (size_t)HDIM * HDIM);
  cast(up_W[1], uwb[1], (size_t)HDIM * HDIM);
  cast(up_W[2], uwb[2], (size_t)HDIM * COUT);
  for (int l = 0; l < 4; ++l)
    transpose_cast_k<<<g1((size_t)GDIM * HDIM, 256), 256, 0, stream>>>(Wih[l], wihT[l],
                                                                       GDIM, HDIM);

  // ---- build adjacency ----
  fill0_k<<<g1((size_t)N0 * N0, 256), 256, 0, stream>>>(Af, (size_t)N0 * N0);
  build_adj_k<<<g1(E, 256), 256, 0, stream>>>(edge, E, Af);
  degdis_k<<<g1(N0, 128), 128, 0, stream>>>(Af, N0, dis[0]);
  cast(Af, Mb[0], (size_t)N0 * N0);

  // ---- first GCN + LSTM (level 0) ----
  xlast_cast_k<<<g1((size_t)N0 * CIN, 256), 256, 0, stream>>>(x, xlb);
  gcn(Mb[0], dis[0], N0, xlb, CIN, dwb[0], HDIM, down_b[0], true, hA);
  dolstm(hA, N0, 0, xs0);
  const float* hptr = xs0;

  // ---- down path: gcn -> lstm -> augment -> topk pool ----
  for (int it = 1; it <= 3; ++it) {
    int lev = it - 1;
    int n = nlev[lev], k = nlev[it];
    cast(hptr, hb, (size_t)n * HDIM);
    gcn(Mb[lev], dis[lev], n, hb, HDIM, dwb[it], HDIM, down_b[it], true, hA);
    dolstm(hA, n, it, hB);
    // augment: M2 = (diag1(M)) @ (diag1(M)); diag removal fused into gather
    set_diag1_k<<<g1((size_t)n * n, 256), 256, 0, stream>>>(Mb[lev], Msb, n);
    gemm(Msb, Msb, Af, n, n, n);
    // topk pooling
    wnorm_k<<<1, HDIM, 0, stream>>>(pool_w[it - 1], invn);
    score_k<<<g1(n, 128), 128, 0, stream>>>(hB, pool_w[it - 1], invn, scoreb, n);
    topk_sort_k<<<1, 1024, (size_t)n * 8, stream>>>(scoreb, n, k, perm[it - 1], sscore);
    float* dest = (it == 1) ? xs1 : (it == 2) ? xs2 : hA;
    gather_x_k<<<g1((size_t)k * HDIM, 256), 256, 0, stream>>>(hB, perm[it - 1], sscore,
                                                              dest, k);
    gather_M_k<<<g1((size_t)k * k, 256), 256, 0, stream>>>(Af, perm[it - 1], n, k, Bf);
    degdis_k<<<g1(k, 128), 128, 0, stream>>>(Bf, k, dis[it]);
    cast(Bf, Mb[it], (size_t)k * k);
    hptr = dest;
  }

  // ---- up path ----
  const float* xsv[3] = {xs0, xs1, xs2};
  const float* hcur = hptr;               // hA, 512 x 128
  for (int i = 0; i < 3; ++i) {
    int j = 2 - i;                        // skip level
    int n = nlev[j], kh = nlev[j + 1];
    int Cout = (i < 2) ? HDIM : COUT;
    (void)hipMemcpyAsync(inbuf, xsv[j], (size_t)n * HDIM * 4,
                         hipMemcpyDeviceToDevice, stream);
    scatter_add_k<<<g1((size_t)kh * HDIM, 256), 256, 0, stream>>>(inbuf, hcur, perm[j],
                                                                  kh, HDIM);
    cast(inbuf, hb, (size_t)n * HDIM);
    float* outp = (i == 0) ? hB : (i == 1) ? hA : (float*)d_out;
    gcn(Mb[j], dis[j], n, hb, HDIM, uwb[i], Cout, up_b[i], i < 2, outp);
    hcur = outp;
  }
}
